// AdapterFusionSentLvlDynamic_11287174054425
// MI455X (gfx1250) — matrix operand order repaired and checked by reading, not compiled, not run on hardware
//
#include <hip/hip_runtime.h>

#ifndef __has_builtin
#define __has_builtin(x) 0
#endif

// ---------------------------------------------------------------------------
// Problem constants: B=8, S=2048, N=8, H=1024, D=64, T=50
// ---------------------------------------------------------------------------
#define BB   8
#define SS   2048
#define NN   8
#define HH   1024
#define DD   64
#define MM   (BB * SS)          // 16384 rows of the big GEMM
#define TINV (1.0f / 50.0f)

typedef __attribute__((ext_vector_type(16))) __bf16          v16bf;
typedef __attribute__((ext_vector_type(8)))  float           v8f;
typedef __attribute__((ext_vector_type(16))) unsigned short  ush16;
typedef __attribute__((ext_vector_type(8)))  unsigned short  ush8;
typedef __attribute__((ext_vector_type(4)))  unsigned short  ush4;
typedef __attribute__((ext_vector_type(4)))  float           f4;
typedef __attribute__((ext_vector_type(4)))  int             v4i;

// ---- gfx1250 async global->LDS path (guarded; falls back to sync copy) ----
#if __has_builtin(__builtin_amdgcn_global_load_async_to_lds_b128)
#define HAVE_ASYNC_LDS 1
// Parameter types per hipcc diagnostic: (v4i AS1*, v4i AS3*, imm int, imm int)
typedef __attribute__((address_space(1))) v4i as1_v4i;
typedef __attribute__((address_space(3))) v4i as3_v4i;
#else
#define HAVE_ASYNC_LDS 0
#endif

#if HAVE_ASYNC_LDS
#if __has_builtin(__builtin_amdgcn_s_wait_asynccnt)
#define WAIT_ASYNC(n) __builtin_amdgcn_s_wait_asynccnt(n)
#else
#define WAIT_ASYNC(n) asm volatile("s_wait_asynccnt " #n ::: "memory")
#endif
#else
#define WAIT_ASYNC(n) do { } while (0)
#endif

__device__ __forceinline__ unsigned short f2bf(float f) {
    unsigned int u = __float_as_uint(f);
    u += 0x7FFFu + ((u >> 16) & 1u);   // round-to-nearest-even
    return (unsigned short)(u >> 16);
}

// ---------------------------------------------------------------------------
// K1: masked mean pooling over S.
// ---------------------------------------------------------------------------
__global__ void pool_kernel(const float* __restrict__ query,
                            const float* __restrict__ key,
                            const int*   __restrict__ amask,
                            float* __restrict__ qsent,
                            float* __restrict__ ksent) {
    const int b = blockIdx.x;
    const int c = blockIdx.y * blockDim.x + threadIdx.x;   // 0..1535
    const int* mrow = amask + b * SS;
    float acc = 0.0f, len = 0.0f;
    if (c < HH) {
        const float* p = query + (size_t)b * SS * HH + c;
        for (int s = 0; s < SS; ++s) {
            float mw = (mrow[s] == 0) ? 1.0f : 0.0f;
            len += mw;
            acc += mw * p[(size_t)s * HH];
        }
        qsent[b * HH + c] = acc / len;
    } else {
        const int cd = c - HH;                              // n*64+d
        const float* p = key + (size_t)b * SS * (NN * DD) + cd;
        for (int s = 0; s < SS; ++s) {
            float mw = (mrow[s] == 0) ? 1.0f : 0.0f;
            len += mw;
            acc += mw * p[(size_t)s * (NN * DD)];
        }
        ksent[b * (NN * DD) + cd] = acc / len;
    }
}

// ---------------------------------------------------------------------------
// K2: per-batch encoders + scores + softmax -> probs[b,n].  grid B, block 256
// ---------------------------------------------------------------------------
__global__ void score_kernel(const float* __restrict__ qsent,
                             const float* __restrict__ ksent,
                             const float* __restrict__ Wq, const float* __restrict__ bq,
                             const float* __restrict__ Wk, const float* __restrict__ bk,
                             float* __restrict__ probs) {
    __shared__ float qenc[DD];
    __shared__ float kenc[NN * DD];
    __shared__ float sc[NN];
    const int b = blockIdx.x;
    const int t = threadIdx.x;

    if (t < DD) {                              // query_enc = qsent @ Wq^T + bq
        float a = bq[t];
        const float* w = Wq + t * HH;
        const float* q = qsent + b * HH;
        for (int h = 0; h < HH; ++h) a += w[h] * q[h];
        qenc[t] = a;
    }
    __syncthreads();
    for (int idx = t; idx < NN * DD; idx += blockDim.x) {   // key_enc
        const int d = idx & (DD - 1);
        const int n = idx >> 6;
        float a = bk[d];
        const float* w  = Wk + d * DD;
        const float* kk = ksent + b * (NN * DD) + n * DD;
        for (int k = 0; k < DD; ++k) a += w[k] * kk[k];
        kenc[idx] = a;
    }
    __syncthreads();
    if (t < NN) {                              // scores[n] = key_enc . query_enc
        float a = 0.0f;
        for (int d = 0; d < DD; ++d) a += kenc[t * DD + d] * qenc[d];
        sc[t] = a;
    }
    __syncthreads();
    if (t == 0) {                              // softmax(scores / T)
        float mx = sc[0];
        for (int n = 1; n < NN; ++n) mx = fmaxf(mx, sc[n]);
        float e[NN], sum = 0.0f;
        for (int n = 0; n < NN; ++n) { e[n] = __expf((sc[n] - mx) * TINV); sum += e[n]; }
        for (int n = 0; n < NN; ++n) probs[b * NN + n] = e[n] / sum;
    }
}

// ---------------------------------------------------------------------------
// K3: Wv fp32 -> bf16 (native [out, in] layout == WMMA B K-contiguous layout)
// ---------------------------------------------------------------------------
__global__ void wv_cvt_kernel(const float* __restrict__ Wv,
                              unsigned short* __restrict__ wvb) {
    const size_t i = ((size_t)blockIdx.x * blockDim.x + threadIdx.x) * 4;
    f4 v = *(const f4*)(Wv + i);
    ush4 r;
#pragma unroll
    for (int j = 0; j < 4; ++j) r[j] = f2bf(v[j]);
    *(ush4*)(wvb + i) = r;
}

// ---------------------------------------------------------------------------
// K4: vbar[m,h] = sum_n probs[b,n] * value[m,n,h]  (fp32 accumulate, bf16 out)
// The mandatory 536 MB HBM stream.  grid MM, block 256, 4 elems/thread.
// ---------------------------------------------------------------------------
__global__ void vbar_kernel(const float* __restrict__ value,
                            const float* __restrict__ probs,
                            unsigned short* __restrict__ vbar) {
    const int m = blockIdx.x;                 // 0..16383 (= b*S + s)
    const int h = threadIdx.x * 4;
    const int b = m >> 11;                    // m / 2048
    const float* vb = value + (size_t)m * NN * HH + h;
    f4 acc = {0.0f, 0.0f, 0.0f, 0.0f};
#pragma unroll
    for (int n = 0; n < NN; ++n) {
        const float p = probs[b * NN + n];
        f4 v = *(const f4*)(vb + (size_t)n * HH);
#pragma unroll
        for (int j = 0; j < 4; ++j) acc[j] += p * v[j];
    }
    ush4 r;
#pragma unroll
    for (int j = 0; j < 4; ++j) r[j] = f2bf(acc[j]);
    *(ush4*)(vbar + (size_t)m * HH + h) = r;
}

// ---------------------------------------------------------------------------
// K5: result = vbar @ Wv^T + bv via V_WMMA_F32_16X16X32_BF16.
// grid (MM/128, HH/64), block 128 = 4 waves.
// Block tile: 128 rows x 64 cols.  Wave w: rows [m0, m0+32), all 4 col tiles
// -> 8 accumulators; A fragments reused 4x, B fragments (from LDS) reused 2x.
// B tile (64 cols x 32 k, 4 KB) is shared by all 4 waves: staged into LDS,
// double-buffered, via async global->LDS copies when available (ASYNCcnt).
//
// VGPR layouts per ISA 7.12.2:
//   A 16x32 bf16: lane<16 -> M=lane, elems K = {kb..kb+7, kb+16..kb+23},
//                 kb = k0 (+8 for lanes>=16).
//   B 32x16 bf16: lane&15 = column, elems K = {kb..kb+7, kb+16..kb+23},
//                 kb = k0 (+8 for lanes>=16), same as A.
//   C/D f32:      VGPR r -> M = r (+8 for lanes>=16), N = lane&15.
// ---------------------------------------------------------------------------
__global__ void __launch_bounds__(128)
gemm_wmma_kernel(const unsigned short* __restrict__ vbar,
                 const unsigned short* __restrict__ wvb,
                 const float* __restrict__ bv,
                 float* __restrict__ out) {
    __shared__ __attribute__((aligned(64))) unsigned short btile[2][64 * 32];

    const int tid   = threadIdx.x;
    const int wave  = tid >> 5;
    const int lane  = tid & 31;
    const int lhalf = lane & 15;
    const bool hi   = lane >= 16;
    const int m0 = (blockIdx.x * 8 + wave * 2) * 16;   // 2 row-tiles per wave
    const int n0 = blockIdx.y * 64;

    // ---- seed accumulators with the bias (sum_n probs == 1) ----
    v8f acc[2][4];
#pragma unroll
    for (int t = 0; t < 4; ++t) {
        const float bias = bv[n0 + t * 16 + lhalf];
#pragma unroll
        for (int r = 0; r < 8; ++r) { acc[0][t][r] = bias; acc[1][t][r] = bias; }
    }

    const unsigned short* aptr0 = vbar + (size_t)(m0 + lhalf) * HH;
    const unsigned short* aptr1 = aptr0 + (size_t)16 * HH;
    const unsigned short* bsrc  = wvb + (size_t)n0 * HH;   // [col][k] block base

    // ---- B tile staging: 256 chunks of 16B; 128 threads x 2 chunks ----
    auto stage = [&](int buf, int k0) {
#pragma unroll
        for (int c = tid; c < 256; c += 128) {
            const int col = c >> 2;            // 0..63
            const int kc  = (c & 3) * 8;       // 0,8,16,24
            const unsigned short* g = bsrc + (size_t)col * HH + k0 + kc;
            unsigned short*       l = &btile[buf][col * 32 + kc];
#if HAVE_ASYNC_LDS
            __builtin_amdgcn_global_load_async_to_lds_b128(
                (as1_v4i*)g, (as3_v4i*)l, 0, 0);
#else
            *(ush8*)l = *(const ush8*)g;
#endif
        }
    };

    stage(0, 0);                                // prime buffer 0

    for (int it = 0; it < HH / 32; ++it) {
        const int k0  = it * 32;
        const int buf = it & 1;

        if (k0 + 32 < HH) {
            __syncthreads();                    // all waves done reading buf^1
            stage(buf ^ 1, k0 + 32);            // prefetch next tile
            WAIT_ASYNC(2);                      // current tile's copies done
        } else {
            WAIT_ASYNC(0);
        }
        __syncthreads();                        // LDS tile visible to block

        // ---- A fragments: 2 row-tiles ----
        const int kb = k0 + (hi ? 8 : 0);
        ush16 apk[2];
        {
            ush8 lo0 = *(const ush8*)(aptr0 + kb);
            ush8 hi0 = *(const ush8*)(aptr0 + kb + 16);
            ush8 lo1 = *(const ush8*)(aptr1 + kb);
            ush8 hi1 = *(const ush8*)(aptr1 + kb + 16);
#pragma unroll
            for (int i = 0; i < 8; ++i) {
                apk[0][i] = lo0[i]; apk[0][i + 8] = hi0[i];
                apk[1][i] = lo1[i]; apk[1][i + 8] = hi1[i];
            }
        }
        const v16bf av0 = __builtin_bit_cast(v16bf, apk[0]);
        const v16bf av1 = __builtin_bit_cast(v16bf, apk[1]);

        // ---- B fragments from LDS + 8 WMMAs ----
        const unsigned short* bl = &btile[buf][hi ? 8 : 0];
#pragma unroll
        for (int t = 0; t < 4; ++t) {
            const unsigned short* bp = bl + (t * 16 + lhalf) * 32;
            ush8 blo = *(const ush8*)(bp);
            ush8 bhi = *(const ush8*)(bp + 16);
            ush16 bpk;
#pragma unroll
            for (int i = 0; i < 8; ++i) { bpk[i] = blo[i]; bpk[i + 8] = bhi[i]; }
            const v16bf bvv = __builtin_bit_cast(v16bf, bpk);
            acc[0][t] = __builtin_amdgcn_wmma_f32_16x16x32_bf16(
                false, av0, false, bvv, (short)0, acc[0][t], false, false);
            acc[1][t] = __builtin_amdgcn_wmma_f32_16x16x32_bf16(
                false, av1, false, bvv, (short)0, acc[1][t], false, false);
        }
    }

    // ---- epilogue: scatter D per the 16x16 f32 C/D VGPR layout ----
#pragma unroll
    for (int r2 = 0; r2 < 2; ++r2) {
#pragma unroll
        for (int t = 0; t < 4; ++t) {
            const int col = n0 + t * 16 + lhalf;
#pragma unroll
            for (int r = 0; r < 8; ++r) {
                const int row = m0 + r2 * 16 + (hi ? 8 + r : r);
                out[(size_t)row * HH + col] = acc[r2][t][r];
            }
        }
    }
}

// ---------------------------------------------------------------------------
extern "C" void kernel_launch(void* const* d_in, const int* in_sizes, int n_in,
                              void* d_out, int out_size, void* d_ws, size_t ws_size,
                              hipStream_t stream) {
    const float* query = (const float*)d_in[0];
    const float* key   = (const float*)d_in[1];
    const float* value = (const float*)d_in[2];
    const int*   amask = (const int*)  d_in[3];
    const float* Wq    = (const float*)d_in[4];
    const float* bq    = (const float*)d_in[5];
    const float* Wk    = (const float*)d_in[6];
    const float* bk    = (const float*)d_in[7];
    const float* Wv    = (const float*)d_in[8];
    const float* bv    = (const float*)d_in[9];
    float* out = (float*)d_out;

    // workspace carve-out (all offsets 256B-aligned)
    char*  ws  = (char*)d_ws;
    size_t off = 0;
    auto carve = [&](size_t bytes) {
        char* p = ws + off;
        off += (bytes + 255) & ~(size_t)255;
        return p;
    };
    float*          probs = (float*)         carve(BB * NN * sizeof(float));
    float*          qsent = (float*)         carve((size_t)BB * HH * sizeof(float));
    float*          ksent = (float*)         carve((size_t)BB * NN * DD * sizeof(float));
    unsigned short* wvb   = (unsigned short*)carve((size_t)HH * HH * sizeof(unsigned short));
    unsigned short* vbar  = (unsigned short*)carve((size_t)MM * HH * sizeof(unsigned short));
    (void)ws_size; (void)in_sizes; (void)n_in; (void)out_size;

    pool_kernel  <<<dim3(BB, (HH + NN * DD) / 256), 256, 0, stream>>>(query, key, amask, qsent, ksent);
    score_kernel <<<BB, 256, 0, stream>>>(qsent, ksent, Wq, bq, Wk, bk, probs);
    wv_cvt_kernel<<<(HH * HH) / (256 * 4), 256, 0, stream>>>(Wv, wvb);
    vbar_kernel  <<<MM, 256, 0, stream>>>(value, probs, vbar);
    gemm_wmma_kernel<<<dim3(MM / 128, HH / 64), 128, 0, stream>>>(vbar, wvb, bv, out);
}
